// AttModel_18743237280566
// MI455X (gfx1250) — matrix operand order repaired and checked by reading, not compiled, not run on hardware
//
#include <hip/hip_runtime.h>
#include <hip/hip_bf16.h>
#include <math.h>

// ---------------------------------------------------------------------------
// MI455X (gfx1250) fused attention:
//   K0: weight transpose+convert  W[k][o] f32 -> WT[o][k] f16   (one-time)
//   K1: qkv projection  x @ W (+bias, relu) -> f16 q/k row-major, f16 v^T
//   K2: flash attention; K/V tiles staged in LDS via async global->LDS copies,
//       online softmax in f32, WMMA for QK^T and PV
//   K3: output projection (+bias, relu) -> f32 d_out
// All matrix math on v_wmma_f32_16x16x32_f16 (wave32).
// ---------------------------------------------------------------------------

typedef __attribute__((ext_vector_type(16))) _Float16 v16h;
typedef __attribute__((ext_vector_type(8)))  _Float16 v8h;
typedef __attribute__((ext_vector_type(8)))  float    v8f;

#define NEGC 9000000000000000.0f

// ----- async global->LDS staging (CDNA5), with compile-safe fallback --------
#if defined(__has_builtin)
#if __has_builtin(__builtin_amdgcn_global_load_async_to_lds_b128)
#define ASYNC_COPY 1
#endif
#endif
#ifndef ASYNC_COPY
#define ASYNC_COPY 0
#endif

#if ASYNC_COPY && defined(__has_builtin) && __has_builtin(__builtin_amdgcn_s_wait_asynccnt)
#define ASYNC_WAIT() __builtin_amdgcn_s_wait_asynccnt(0)
#elif ASYNC_COPY
#define ASYNC_WAIT() asm volatile("s_wait_asynccnt 0" ::: "memory")
#else
#define ASYNC_WAIT()
#endif

// Builtin signature (from hipcc diagnostic): param0 is
// 'int __vector(4) __device__ *' -> gcc vector_size(16) int in addrspace(1).
typedef int b128 __attribute__((vector_size(16)));
typedef __attribute__((address_space(1))) b128 gb128;
typedef __attribute__((address_space(3))) b128 lb128;

__device__ __forceinline__ void async_cp16(const void* g, void* l) {
#if ASYNC_COPY
  __builtin_amdgcn_global_load_async_to_lds_b128(
      (gb128*)(uintptr_t)g,
      (lb128*)(unsigned)(uintptr_t)l, 0, 0);
#else
  *(v8h*)l = *(const v8h*)g;
#endif
}

// Copy nrows rows of 64 bytes (global row stride gstride bytes) into packed
// LDS (64B rows). 256 threads, 16B per lane per async op.
__device__ __forceinline__ void stage_rows64(const char* g, size_t gstride,
                                             char* l, int nrows, int tid) {
  const int total = nrows * 64;
  for (int f = tid * 16; f < total; f += 4096) {
    const int r = f >> 6, w = f & 63;
    async_cp16(g + (size_t)r * gstride + w, l + f);
  }
}

// ----- WMMA fragment helpers ------------------------------------------------
__device__ __forceinline__ v8f zero8() {
  v8f z;
#pragma unroll
  for (int i = 0; i < 8; ++i) z[i] = 0.0f;
  return z;
}

__device__ __forceinline__ v16h combine16(v8h lo, v8h hi) {
  v16h r;
#pragma unroll
  for (int i = 0; i < 8; ++i) { r[i] = lo[i]; r[i + 8] = hi[i]; }
  return r;
}

// A-fragment (16x32 f16). rowk = this lane's row (m = lane&15) at k-chunk base.
__device__ __forceinline__ v16h load_a_f16(const _Float16* rowk, int lane) {
  const int koff = (lane & 16) ? 8 : 0;
  v8h lo = *(const v8h*)(rowk + koff);
  v8h hi = *(const v8h*)(rowk + 16 + koff);
  return combine16(lo, hi);
}

__device__ __forceinline__ v16h load_a_f32cvt(const float* rowk, int lane) {
  const int koff = (lane & 16) ? 8 : 0;
  v16h r;
#pragma unroll
  for (int i = 0; i < 8; ++i) {
    r[i]     = (_Float16)rowk[koff + i];
    r[i + 8] = (_Float16)rowk[16 + koff + i];
  }
  return r;
}

// B-fragment (32x16 f16). colbase = K=0 of this lane's column (n = lane&15),
// 32 K values contiguous. Lanes 0-15: K=0..7,16..23, lanes 16-31: K=8..15,24..31.
__device__ __forceinline__ v16h load_b_f16(const _Float16* colbase, int lane) {
  const int koff = (lane & 16) ? 8 : 0;
  v8h lo = *(const v8h*)(colbase + koff);
  v8h hi = *(const v8h*)(colbase + 16 + koff);
  return combine16(lo, hi);
}

__device__ __forceinline__ v8f wmma_f16(v16h a, v16h b, v8f c) {
  return __builtin_amdgcn_wmma_f32_16x16x32_f16(false, a, false, b,
                                                (short)0, c, false, false);
}

// ---------------------------------------------------------------------------
// K0: weight transpose + f16 convert. grid = (256, 4), block = 256.
// ---------------------------------------------------------------------------
__global__ __launch_bounds__(256) void wtrans_kernel(
    const float* __restrict__ Wq, const float* __restrict__ Wk,
    const float* __restrict__ Wv, const float* __restrict__ Wo,
    _Float16* __restrict__ WqT, _Float16* __restrict__ WkT,
    _Float16* __restrict__ WvT, _Float16* __restrict__ WoT) {
  const int o = blockIdx.x, k = threadIdx.x, s = blockIdx.y;
  const float* src = (s == 0) ? Wq : (s == 1) ? Wk : (s == 2) ? Wv : Wo;
  _Float16*    dst = (s == 0) ? WqT : (s == 1) ? WkT : (s == 2) ? WvT : WoT;
  dst[(size_t)o * 256 + k] = (_Float16)src[(size_t)k * 256 + o];
}

// ---------------------------------------------------------------------------
// K1: QKV projection. grid = (TOK/64, 3), block = 256 (8 waves).
// blockIdx.y selects WqT/WkT/WvT. Each wave: 16 rows x 128 cols (8 sub-tiles).
// ---------------------------------------------------------------------------
__global__ __launch_bounds__(256) void qkv_proj_kernel(
    const float* __restrict__ x,
    const _Float16* __restrict__ WqT, const float* __restrict__ bq,
    const _Float16* __restrict__ WkT, const float* __restrict__ bk,
    const _Float16* __restrict__ WvT, const float* __restrict__ bv,
    _Float16* __restrict__ q16, _Float16* __restrict__ k16,
    _Float16* __restrict__ vT) {
  __shared__ alignas(32) _Float16 Wt[256 * 32];  // W chunk [col][k] (16 KB)

  const int lane = threadIdx.x & 31;
  const int wave = threadIdx.x >> 5;
  const int nlo  = lane & 15;
  const int which = blockIdx.y;
  const _Float16* WT  = (which == 0) ? WqT : (which == 1) ? WkT : WvT;
  const float*   bias = (which == 0) ? bq  : (which == 1) ? bk  : bv;

  const int g0 = blockIdx.x * 64;
  const int rt = wave & 3;
  const int ct = wave >> 2;
  const int row_lane = g0 + rt * 16 + nlo;

  v8f acc[8];
#pragma unroll
  for (int t = 0; t < 8; ++t) acc[t] = zero8();

  for (int kk = 0; kk < 256; kk += 32) {
    // Stage WT[0:256 cols][kk:kk+32] -> LDS (async, 64B per col).
    stage_rows64((const char*)WT + kk * 2, 512, (char*)Wt, 256, threadIdx.x);
    ASYNC_WAIT();
    __syncthreads();

    v16h a = load_a_f32cvt(x + (size_t)row_lane * 256 + kk, lane);
#pragma unroll
    for (int t = 0; t < 8; ++t) {
      const int c = ct * 128 + t * 16 + nlo;
      acc[t] = wmma_f16(a, load_b_f16(Wt + c * 32, lane), acc[t]);
    }
    __syncthreads();
  }

  const int b   = g0 >> 9;
  const int tn0 = (g0 & 511) + rt * 16;
#pragma unroll
  for (int t = 0; t < 8; ++t) {
    const int n = ct * 128 + t * 16 + nlo;
    const float bi = bias[n];
#pragma unroll
    for (int r = 0; r < 8; ++r) {
      const int m = (lane & 16) ? (r + 8) : r;
      const float val = fmaxf(acc[t][r] + bi, 0.0f);
      if (which == 2) {
        vT[((size_t)b * 256 + n) * 512 + (tn0 + m)] = (_Float16)val;
      } else {
        _Float16* dst = (which == 0) ? q16 : k16;
        dst[(size_t)(g0 + rt * 16 + m) * 256 + n] = (_Float16)val;
      }
    }
  }
}

// ---------------------------------------------------------------------------
// K2: flash attention. grid = TOK/128, block = 256 (8 waves x 16 q rows).
// K/V tiles staged once per block in LDS (async); S and P never hit HBM.
// ---------------------------------------------------------------------------
__global__ __launch_bounds__(256) void attn_kernel(
    const float* __restrict__ mask,
    const _Float16* __restrict__ q16, const _Float16* __restrict__ k16,
    const _Float16* __restrict__ vT, _Float16* __restrict__ o16) {
  __shared__ alignas(32) _Float16 Kt[32 * 256];   // [key][h]   16 KB
  __shared__ alignas(32) _Float16 Vt[256 * 32];   // [h][key]   16 KB
  __shared__ alignas(32) _Float16 Pl[8][16 * 32]; // per-wave P bounce 8 KB

  const int tid  = threadIdx.x;
  const int lane = tid & 31;
  const int wave = tid >> 5;
  const int nlo  = lane & 15;
  const int g0   = blockIdx.x * 128 + wave * 16;  // first global q row of wave
  const int b    = g0 >> 9;
  const int qr0  = g0 & 511;
  const size_t qrow = (size_t)(g0 + nlo) * 256;
  const char* kB = (const char*)(k16 + (size_t)b * 512 * 256);
  const char* vB = (const char*)(vT + (size_t)b * 256 * 512);

  v8f O[16];
#pragma unroll
  for (int t = 0; t < 16; ++t) O[t] = zero8();
  float mi[8], li[8];
#pragma unroll
  for (int r = 0; r < 8; ++r) { mi[r] = -INFINITY; li[r] = 0.0f; }

  _Float16* pw = Pl[wave];

  for (int j = 0; j < 512; j += 32) {
    if (j) __syncthreads();  // previous tiles fully consumed
    // Stage K[j:j+32][0:256] (linear) and V^T[0:256][j:j+32] (64B rows).
    stage_rows64(kB + (size_t)j * 512, 64, (char*)Kt, 256, tid);
    stage_rows64(vB + (size_t)j * 2, 1024, (char*)Vt, 256, tid);
    ASYNC_WAIT();
    __syncthreads();

    if (j + 32 < 512)
      __builtin_prefetch(mask + ((size_t)b * 512 + qr0 + nlo) * 512 + j + 32, 0, 1);

    // S = q(16x256) @ k(32x256)^T -> two 16x16 f32 tiles (from LDS K tile)
    v8f s0 = zero8(), s1 = zero8();
#pragma unroll
    for (int kc = 0; kc < 8; ++kc) {
      v16h a = load_a_f16(q16 + qrow + kc * 32, lane);
      s0 = wmma_f16(a, load_b_f16(Kt + (nlo)      * 256 + kc * 32, lane), s0);
      s1 = wmma_f16(a, load_b_f16(Kt + (16 + nlo) * 256 + kc * 32, lane), s1);
    }

    // Mask + online softmax per q row (rows live across 16-lane half-waves).
    float p0[8], p1[8], scl[8];
#pragma unroll
    for (int r = 0; r < 8; ++r) {
      const int m = (lane & 16) ? (r + 8) : r;
      const float* mrow = mask + ((size_t)b * 512 + qr0 + m) * 512 + j;
      const float mv0 = mrow[nlo], mv1 = mrow[16 + nlo];
      float a0 = s0[r] * mv0 - NEGC * (1.0f - mv0);
      float a1 = s1[r] * mv1 - NEGC * (1.0f - mv1);
      float mx = fmaxf(a0, a1);
#pragma unroll
      for (int o = 1; o < 16; o <<= 1) mx = fmaxf(mx, __shfl_xor(mx, o, 32));
      const float nm = fmaxf(mi[r], mx);
      const float sc = __expf(mi[r] - nm);
      const float e0 = __expf(a0 - nm);
      const float e1 = __expf(a1 - nm);
      float rs = e0 + e1;
#pragma unroll
      for (int o = 1; o < 16; o <<= 1) rs += __shfl_xor(rs, o, 32);
      li[r] = li[r] * sc + rs;
      mi[r] = nm;
      p0[r] = e0; p1[r] = e1; scl[r] = sc;
    }
#pragma unroll
    for (int t = 0; t < 16; ++t)
#pragma unroll
      for (int r = 0; r < 8; ++r) O[t][r] *= scl[r];

    // C-layout -> A-layout transpose of P through per-wave LDS.
#pragma unroll
    for (int r = 0; r < 8; ++r) {
      const int m = (lane & 16) ? (r + 8) : r;
      pw[m * 32 + nlo]      = (_Float16)p0[r];
      pw[m * 32 + 16 + nlo] = (_Float16)p1[r];
    }
    asm volatile("s_wait_dscnt 0" ::: "memory");  // cross-lane LDS RAW fence
    v16h pa = load_a_f16(pw + nlo * 32, lane);

    // O += P(16x32) @ V(32x256) from LDS V tile (two 16B LDS reads per B-fragment).
#pragma unroll
    for (int t = 0; t < 16; ++t)
      O[t] = wmma_f16(pa, load_b_f16(Vt + (t * 16 + nlo) * 32, lane), O[t]);
  }

  // Normalize and store o16 row-major f16.
#pragma unroll
  for (int r = 0; r < 8; ++r) {
    const float inv = 1.0f / li[r];
    const int m = (lane & 16) ? (r + 8) : r;
#pragma unroll
    for (int t = 0; t < 16; ++t)
      o16[(size_t)(g0 + m) * 256 + t * 16 + nlo] = (_Float16)(O[t][r] * inv);
  }
}

// ---------------------------------------------------------------------------
// K3: output projection. grid = TOK/64, block = 256 (8 waves). f32 out + relu.
// ---------------------------------------------------------------------------
__global__ __launch_bounds__(256) void out_proj_kernel(
    const _Float16* __restrict__ o16,
    const _Float16* __restrict__ WoT, const float* __restrict__ bo,
    float* __restrict__ out) {
  __shared__ alignas(32) _Float16 Wt[256 * 32];

  const int lane = threadIdx.x & 31;
  const int wave = threadIdx.x >> 5;
  const int nlo  = lane & 15;
  const int g0 = blockIdx.x * 64;
  const int rt = wave & 3;
  const int ct = wave >> 2;
  const int row_lane = g0 + rt * 16 + nlo;

  v8f acc[8];
#pragma unroll
  for (int t = 0; t < 8; ++t) acc[t] = zero8();

  for (int kk = 0; kk < 256; kk += 32) {
    stage_rows64((const char*)WoT + kk * 2, 512, (char*)Wt, 256, threadIdx.x);
    ASYNC_WAIT();
    __syncthreads();

    v16h a = load_a_f16(o16 + (size_t)row_lane * 256 + kk, lane);
#pragma unroll
    for (int t = 0; t < 8; ++t) {
      const int c = ct * 128 + t * 16 + nlo;
      acc[t] = wmma_f16(a, load_b_f16(Wt + c * 32, lane), acc[t]);
    }
    __syncthreads();
  }

#pragma unroll
  for (int t = 0; t < 8; ++t) {
    const int n = ct * 128 + t * 16 + nlo;
    const float bi = bo[n];
#pragma unroll
    for (int r = 0; r < 8; ++r) {
      const int m = (lane & 16) ? (r + 8) : r;
      out[(size_t)(g0 + rt * 16 + m) * 256 + n] = fmaxf(acc[t][r] + bi, 0.0f);
    }
  }
}

// ---------------------------------------------------------------------------
extern "C" void kernel_launch(void* const* d_in, const int* in_sizes, int n_in,
                              void* d_out, int out_size, void* d_ws,
                              size_t ws_size, hipStream_t stream) {
  (void)in_sizes; (void)n_in; (void)out_size; (void)ws_size;
  // setup_inputs order: x, mask, Wv, bv, Wk, bk, Wq, bq, Wo, bo
  const float* x    = (const float*)d_in[0];
  const float* mask = (const float*)d_in[1];
  const float* Wv   = (const float*)d_in[2];
  const float* bv   = (const float*)d_in[3];
  const float* Wk   = (const float*)d_in[4];
  const float* bk   = (const float*)d_in[5];
  const float* Wq   = (const float*)d_in[6];
  const float* bq   = (const float*)d_in[7];
  const float* Wo   = (const float*)d_in[8];
  const float* bo   = (const float*)d_in[9];
  float* out = (float*)d_out;

  const size_t TOK = 128u * 512u;  // B*N
  _Float16* q16 = (_Float16*)d_ws;         // 32 MB
  _Float16* k16 = q16 + TOK * 256;         // 32 MB
  _Float16* vT  = k16 + TOK * 256;         // 32 MB, [B,H,N]
  _Float16* o16 = vT  + TOK * 256;         // 32 MB
  _Float16* WqT = o16 + TOK * 256;         // 128 KB each, [o][k] f16
  _Float16* WkT = WqT + 256 * 256;
  _Float16* WvT = WkT + 256 * 256;
  _Float16* WoT = WvT + 256 * 256;

  wtrans_kernel<<<dim3(256, 4), 256, 0, stream>>>(Wq, Wk, Wv, Wo,
                                                  WqT, WkT, WvT, WoT);
  qkv_proj_kernel<<<dim3((unsigned)(TOK / 64), 3), 256, 0, stream>>>(
      x, WqT, bq, WkT, bk, WvT, bv, q16, k16, vT);
  attn_kernel<<<dim3((unsigned)(TOK / 128)), 256, 0, stream>>>(
      mask, q16, k16, vT, o16);
  out_proj_kernel<<<dim3((unsigned)(TOK / 64)), 256, 0, stream>>>(
      o16, WoT, bo, out);
}
